// GNNConv_85607288144369
// MI455X (gfx1250) — matrix operand order repaired and checked by reading, not compiled, not run on hardware
//
#include <hip/hip_runtime.h>

typedef __attribute__((ext_vector_type(16))) _Float16 v16h;
typedef __attribute__((ext_vector_type(8)))  _Float16 v8h;
typedef __attribute__((ext_vector_type(8)))  float    v8f;
typedef __attribute__((ext_vector_type(4)))  unsigned int v4u;
typedef __attribute__((ext_vector_type(8)))  int      v8i;
typedef __attribute__((ext_vector_type(4)))  int      v4i;

#define DFEAT 512
#define BM 32
#define BK 32
#define LN_EPS 1e-5f

// ---------------------------------------------------------------- utilities
__global__ void zero_f_kernel(float* __restrict__ p, long n) {
    long i = (long)blockIdx.x * blockDim.x + threadIdx.x;
    if (i < n) p[i] = 0.0f;
}

__global__ void count_deg_kernel(const int* __restrict__ tgt, float* __restrict__ cnt, int E) {
    int i = blockIdx.x * blockDim.x + threadIdx.x;
    if (i < E) {
        __hip_atomic_fetch_add(&cnt[tgt[i]], 1.0f, __ATOMIC_RELAXED, __HIP_MEMORY_SCOPE_AGENT);
    }
}

__global__ void invert_cnt_kernel(float* __restrict__ cnt, int N) {
    int i = blockIdx.x * blockDim.x + threadIdx.x;
    if (i < N) cnt[i] = 1.0f / fmaxf(cnt[i], 1.0f);
}

// f32 -> f16 bulk convert (8 elements / thread, 16B stores)
__global__ void cvt_f16_kernel(const float* __restrict__ in, _Float16* __restrict__ outp, long n8) {
    long i = (long)blockIdx.x * blockDim.x + threadIdx.x;
    if (i >= n8) return;
    const float4* p = (const float4*)(in + i * 8);
    float4 a = p[0], b = p[1];
    v8h h = {(_Float16)a.x, (_Float16)a.y, (_Float16)a.z, (_Float16)a.w,
             (_Float16)b.x, (_Float16)b.y, (_Float16)b.z, (_Float16)b.w};
    *(v8h*)(outp + i * 8) = h;
}

// agg[tgt] += x[src] * inv_cnt[tgt]  (fused mean aggregation, f32 atomics)
__global__ void scatter_mean_kernel(const float* __restrict__ x,
                                    const int* __restrict__ src,
                                    const int* __restrict__ tgt,
                                    const float* __restrict__ inv,
                                    float* __restrict__ agg, int E) {
    long tid = (long)blockIdx.x * blockDim.x + threadIdx.x;
    int e = (int)(tid >> 7);
    if (e >= E) return;
    int c = ((int)tid & 127) << 2;
    int s = src[e];
    int t = tgt[e];
    float ic = inv[t];
    const float4 v = *(const float4*)(x + (long)s * DFEAT + c);
    float* dst = agg + (long)t * DFEAT + c;
    __hip_atomic_fetch_add(dst + 0, v.x * ic, __ATOMIC_RELAXED, __HIP_MEMORY_SCOPE_AGENT);
    __hip_atomic_fetch_add(dst + 1, v.y * ic, __ATOMIC_RELAXED, __HIP_MEMORY_SCOPE_AGENT);
    __hip_atomic_fetch_add(dst + 2, v.z * ic, __ATOMIC_RELAXED, __HIP_MEMORY_SCOPE_AGENT);
    __hip_atomic_fetch_add(dst + 3, v.w * ic, __ATOMIC_RELAXED, __HIP_MEMORY_SCOPE_AGENT);
}

// ---------------------------------------------- Tensor Data Mover (D#) load
// 2-D tile load: tile (tile1 rows x tile0 elems) of an f16 tensor with row
// stride stride0 -> packed contiguously at LDS byte offset lds_off.
// D# bitfields per CDNA5 ISA:
//   g0: count=1 | lds_addr | global_addr[56:0] | type=2
//   g1: data_size=1(2B) | tensor_dim0/1 | tile_dim0/1 (tile_dim2=0) | dim0_stride
// groups 2/3 zero (2-D tensor; tile_dim2=0 -> unused).
// This toolchain exposes the 6-arg builtin:
//   (uint32x4 g0, int32x8 g1, int32x4, int32x4, int32x8, i32 cpol)
__device__ __forceinline__ void tdm_load_2d_f16(unsigned int lds_off, const void* gptr,
                                                unsigned int dim0, unsigned int dim1,
                                                unsigned int tile0, unsigned int tile1,
                                                unsigned int stride0) {
    unsigned long long ga = (unsigned long long)(size_t)gptr;
    v4u g0;
    g0[0] = 1u;                                              // count=1
    g0[1] = lds_off;                                         // lds_addr (bytes)
    g0[2] = (unsigned int)(ga & 0xffffffffull);              // global_addr[31:0]
    g0[3] = (unsigned int)((ga >> 32) & 0x1ffffffull) | 0x80000000u; // [56:32] | type=2
    v8i g1;
    g1[0] = (int)(1u << 16);                                 // data_size = 2 bytes
    g1[1] = (int)((dim0 & 0xffffu) << 16);                   // tensor_dim0[15:0]
    g1[2] = (int)((dim0 >> 16) | ((dim1 & 0xffffu) << 16));  // dim0[31:16] | dim1[15:0]
    g1[3] = (int)((dim1 >> 16) | ((tile0 & 0xffffu) << 16)); // dim1[31:16] | tile_dim0
    g1[4] = (int)(tile1 & 0xffffu);                          // tile_dim1 (tile_dim2 = 0)
    g1[5] = (int)stride0;                                    // tensor_dim0_stride[31:0]
    g1[6] = 0;                                               // stride0[47:32] | stride1 lo
    g1[7] = 0;
    v4i z4 = {0, 0, 0, 0};
    v8i z8 = {0, 0, 0, 0, 0, 0, 0, 0};
    __builtin_amdgcn_tensor_load_to_lds(g0, g1, z4, z4, z8, 0);
}

// -------------------------------------------------- fused SAGE layer kernel
// out = relu(LN(agg @ Wl^T + x @ Wr^T + b)) + x     for a 32-row block.
// Operands pre-converted to f16; A/B tiles DMA'd to LDS by the TDM, double
// buffered (TENSORcnt), 8 waves: 2 row-groups x 4 col-groups of 128.
__global__ __launch_bounds__(256)
void sage_layer_kernel(const _Float16* __restrict__ aggh, const _Float16* __restrict__ xh,
                       const float* __restrict__ xin,
                       const _Float16* __restrict__ Wlh, const _Float16* __restrict__ Wrh,
                       const float* __restrict__ bias, const float* __restrict__ lng,
                       const float* __restrict__ lnb, float* __restrict__ out, int N) {
    __shared__ __align__(32) union {
        struct {
            _Float16 A[BM * BK];     //  2 KB : 32 rows x 32 K
            _Float16 B[DFEAT * BK];  // 32 KB : Bs[n][k] = W[n][k0+k]
        } stage[2];                  // double buffered (68 KB)
        float h[BM * DFEAT];         // 64 KB : epilogue h tile
    } sm;

    const int tid   = threadIdx.x;
    const int lane  = tid & 31;
    const int w     = tid >> 5;
    const int wr    = w >> 2;      // 0..1 : 16-row group
    const int wc    = w & 3;       // 0..3 : 128-col group
    const int m16   = lane & 15;
    const int khalf = lane >> 4;
    const int row0  = blockIdx.x * BM;
    const int S     = 2 * (DFEAT / BK);   // 32 k-steps over [agg|x]

    v8f acc[8] = {};

    // issue TDM for step s into buffer (s & 1)
    auto issue = [&](int s) {
        int ph = s >> 4;
        unsigned int k0 = (unsigned int)(s & 15) * BK;
        const _Float16* Asrc = ph ? xh : aggh;
        const _Float16* Bsrc = ph ? Wrh : Wlh;
        unsigned int ldsA = (unsigned int)(size_t)&sm.stage[s & 1].A[0];
        unsigned int ldsB = (unsigned int)(size_t)&sm.stage[s & 1].B[0];
        // A tile: BM rows x BK k, row stride DFEAT; OOB rows (dim1) read zero
        tdm_load_2d_f16(ldsA, Asrc + (size_t)row0 * DFEAT + k0,
                        DFEAT - k0, (unsigned int)(N - row0), BK, BM, DFEAT);
        // B tile: DFEAT rows x BK k of W
        tdm_load_2d_f16(ldsB, Bsrc + k0,
                        DFEAT - k0, DFEAT, BK, DFEAT, DFEAT);
    };

    if (w == 0) issue(0);

    for (int s = 0; s < S; ++s) {
        if (w == 0) {
            if (s + 1 < S) {
                issue(s + 1);                        // prefetch next buffer
                __builtin_amdgcn_s_wait_tensorcnt(2); // current buffer's pair done
            } else {
                __builtin_amdgcn_s_wait_tensorcnt(0);
            }
        }
        __syncthreads();

        const _Float16* As = sm.stage[s & 1].A;
        const _Float16* Bs = sm.stage[s & 1].B;

        // A fragment (16x32 f16): lane = row M=m16,
        // K = khalf*16 + {0..15}   (contiguous, matches B)
        const int am = wr * 16 + m16;
        v16h a = *(const v16h*)&As[am * BK + khalf * 16];

        #pragma unroll
        for (int t = 0; t < 8; ++t) {
            int n = wc * 128 + t * 16 + m16;
            v16h bf = *(const v16h*)&Bs[n * BK + khalf * 16];  // K = khalf*16 + h
            acc[t] = __builtin_amdgcn_wmma_f32_16x16x32_f16(
                false, a, false, bf, (short)0, acc[t], false, false);
        }
        __syncthreads();
    }

    // ---- epilogue: h = acc + bias -> LDS (overlaps stage buffers; all waves
    //      passed the final barrier after the last WMMA reads)
    #pragma unroll
    for (int t = 0; t < 8; ++t) {
        int col = wc * 128 + t * 16 + m16;
        float bv = bias[col];
        #pragma unroll
        for (int r = 0; r < 8; ++r) {
            int row = wr * 16 + khalf * 8 + r;   // D layout: M = r + 8*khalf
            sm.h[row * DFEAT + col] = acc[t][r] + bv;
        }
    }
    __syncthreads();

    // ---- LayerNorm + ReLU + residual: 8 threads per row, 64 elems each
    {
        int row = tid >> 3;
        int seg = tid & 7;
        int node = row0 + row;
        const float* hrow = &sm.h[row * DFEAT];
        float s = 0.f, ss = 0.f;
        #pragma unroll
        for (int c4 = seg * 16; c4 < seg * 16 + 16; ++c4) {
            float4 hv = *(const float4*)&hrow[c4 * 4];
            s  += hv.x + hv.y + hv.z + hv.w;
            ss += hv.x * hv.x + hv.y * hv.y + hv.z * hv.z + hv.w * hv.w;
        }
        #pragma unroll
        for (int m = 1; m < 8; m <<= 1) {
            s  += __shfl_xor(s,  m, 8);
            ss += __shfl_xor(ss, m, 8);
        }
        float mu  = s * (1.0f / DFEAT);
        float var = ss * (1.0f / DFEAT) - mu * mu;
        float rs  = rsqrtf(var + LN_EPS);
        if (node < N) {
            const float* xr = xin + (long)node * DFEAT;
            float* orow     = out + (long)node * DFEAT;
            #pragma unroll
            for (int c4 = seg * 16; c4 < seg * 16 + 16; ++c4) {
                float4 hv = *(const float4*)&hrow[c4 * 4];
                float4 gv = *(const float4*)&lng[c4 * 4];
                float4 bv = *(const float4*)&lnb[c4 * 4];
                float4 xv = *(const float4*)&xr[c4 * 4];
                float4 ov;
                ov.x = fmaxf((hv.x - mu) * rs * gv.x + bv.x, 0.f) + xv.x;
                ov.y = fmaxf((hv.y - mu) * rs * gv.y + bv.y, 0.f) + xv.y;
                ov.z = fmaxf((hv.z - mu) * rs * gv.z + bv.z, 0.f) + xv.z;
                ov.w = fmaxf((hv.w - mu) * rs * gv.w + bv.w, 0.f) + xv.w;
                *(float4*)&orow[c4 * 4] = ov;
            }
        }
    }
}

// ------------------------------------------------------------------ launch
extern "C" void kernel_launch(void* const* d_in, const int* in_sizes, int n_in,
                              void* d_out, int out_size, void* d_ws, size_t ws_size,
                              hipStream_t stream) {
    const float* x   = (const float*)d_in[0];
    const int*   ei  = (const int*)d_in[1];
    const float* W_l = (const float*)d_in[2];
    const float* W_r = (const float*)d_in[3];
    const float* b   = (const float*)d_in[4];
    const float* lng = (const float*)d_in[5];
    const float* lnb = (const float*)d_in[6];

    const int N = in_sizes[0] / DFEAT;
    const int E = in_sizes[1] / 2;
    const int L = in_sizes[4] / DFEAT;
    const int* src = ei;
    const int* tgt = ei + E;
    const long ND = (long)N * DFEAT;
    const long DD = (long)DFEAT * DFEAT;

    // workspace: inv | agg(f32) | bufA | bufB | aggh(f16) | xh(f16) | Wh(f16)
    char* wsp = (char*)d_ws;
    size_t off = 0;
    auto carve = [&](size_t bytes) {
        char* p = wsp + off;
        off += (bytes + 255) & ~(size_t)255;
        return p;
    };
    float*    inv  = (float*)carve((size_t)N * 4);
    float*    agg  = (float*)carve((size_t)ND * 4);
    float*    bufA = (float*)carve((size_t)ND * 4);
    float*    bufB = (float*)carve((size_t)ND * 4);
    _Float16* aggh = (_Float16*)carve((size_t)ND * 2);
    _Float16* xh   = (_Float16*)carve((size_t)ND * 2);
    _Float16* Wh   = (_Float16*)carve((size_t)L * 2 * DD * 2);

    // degree -> inv_cnt (once)
    zero_f_kernel<<<(N + 255) / 256, 256, 0, stream>>>(inv, N);
    count_deg_kernel<<<(E + 255) / 256, 256, 0, stream>>>(tgt, inv, E);
    invert_cnt_kernel<<<(N + 255) / 256, 256, 0, stream>>>(inv, N);

    // pre-convert all layer weights to f16 (once)
    for (int l = 0; l < L; ++l) {
        long n8 = DD / 8;
        cvt_f16_kernel<<<(int)((n8 + 255) / 256), 256, 0, stream>>>(
            W_l + (size_t)l * DD, Wh + (size_t)(2 * l) * DD, n8);
        cvt_f16_kernel<<<(int)((n8 + 255) / 256), 256, 0, stream>>>(
            W_r + (size_t)l * DD, Wh + (size_t)(2 * l + 1) * DD, n8);
    }

    const float* cur = x;
    for (int l = 0; l < L; ++l) {
        zero_f_kernel<<<(int)((ND + 255) / 256), 256, 0, stream>>>(agg, ND);
        long sthreads = (long)E * (DFEAT / 4);
        scatter_mean_kernel<<<(int)((sthreads + 255) / 256), 256, 0, stream>>>(
            cur, src, tgt, inv, agg, E);
        // f16 copies of this layer's GEMM operands
        long n8 = ND / 8;
        cvt_f16_kernel<<<(int)((n8 + 255) / 256), 256, 0, stream>>>(agg, aggh, n8);
        cvt_f16_kernel<<<(int)((n8 + 255) / 256), 256, 0, stream>>>(cur, xh, n8);

        float* outp = (l == L - 1) ? (float*)d_out : ((l & 1) ? bufB : bufA);
        sage_layer_kernel<<<(N + BM - 1) / BM, 256, 0, stream>>>(
            aggh, xh, cur,
            Wh + (size_t)(2 * l) * DD, Wh + (size_t)(2 * l + 1) * DD,
            b + (size_t)l * DFEAT, lng + (size_t)l * DFEAT, lnb + (size_t)l * DFEAT,
            outp, N);
        cur = outp;
    }
}
